// LatentAttentionHead_87660282512109
// MI455X (gfx1250) — matrix operand order repaired and checked by reading, not compiled, not run on hardware
//
#include <hip/hip_runtime.h>
#include <hip/hip_bf16.h>
#include <stdint.h>

// ---------------------------------------------------------------------------
// Problem sizes (fixed by the reference)
// ---------------------------------------------------------------------------
#define EMBED 2048
#define HDIM  128
#define LDIM  512
#define NLAT  2048
#define SEQ   2048
#define NBATCH 4

// ---------------------------------------------------------------------------
// WMMA types (CDNA5 / gfx1250, wave32)
// ---------------------------------------------------------------------------
typedef __attribute__((ext_vector_type(16))) __bf16        v16bf;
typedef __attribute__((ext_vector_type(8)))  float         v8f;
typedef __attribute__((ext_vector_type(4)))  unsigned int  v4u;

union FragBF {
    v16bf bf;
    v4u   u[2];
};

__device__ __forceinline__ unsigned short f2bf(float f) {
    unsigned int u = __float_as_uint(f);
    u += 0x7FFFu + ((u >> 16) & 1u);          // round-to-nearest-even
    return (unsigned short)(u >> 16);
}

__device__ __forceinline__ v4u ld16B(const unsigned short* p) {
    return *reinterpret_cast<const v4u*>(p);
}

__device__ __forceinline__ v8f wmma_bf16(const FragBF& a, const FragBF& b, v8f c) {
    return __builtin_amdgcn_wmma_f32_16x16x32_bf16(false, a.bf, false, b.bf,
                                                   (short)0, c, false, false);
}

// ---------------------------------------------------------------------------
// Phase 0: fp32 -> bf16 conversion (packs 2 per dword)
// ---------------------------------------------------------------------------
__global__ void cvt_bf16_kernel(const float* __restrict__ src,
                                unsigned int* __restrict__ dst, int npairs) {
    int i = blockIdx.x * blockDim.x + threadIdx.x;
    if (i >= npairs) return;
    float a = src[2 * i];
    float b = src[2 * i + 1];
    dst[i] = (unsigned int)f2bf(a) | ((unsigned int)f2bf(b) << 16);
}

// ---------------------------------------------------------------------------
// Phase 1: Q = latents @ Wq^T   [NLAT x HDIM], K = LDIM. One wave = 16x16 tile.
// NT GEMM: both operands row-major with contiguous K.
// ---------------------------------------------------------------------------
__global__ void __launch_bounds__(256)
qproj_kernel(const unsigned short* __restrict__ lat,
             const unsigned short* __restrict__ wq,
             unsigned short* __restrict__ qout) {
    const int lane = threadIdx.x & 31;
    const int wave = threadIdx.x >> 5;
    const int tile = blockIdx.x * 8 + wave;         // 1024 tiles total
    const int mt = tile >> 3;                       // 0..127  (n rows)
    const int ht = tile & 7;                        // 0..7    (h cols)
    const int m16 = lane & 15;
    const int hi  = lane >> 4;

    v8f acc = {};
    const unsigned short* arow = lat + (size_t)(mt * 16 + m16) * LDIM;
    const unsigned short* brow = wq  + (size_t)(ht * 16 + m16) * LDIM;

    for (int k = 0; k < LDIM; k += 32) {
        FragBF a, b;
        // A frag: lane<16 holds K k..k+7 (u0), k+16..k+23 (u1); lane>=16: +8
        a.u[0] = ld16B(arow + k + hi * 8);
        a.u[1] = ld16B(arow + k + 16 + hi * 8);
        // B frag: same K layout as A
        b.u[0] = ld16B(brow + k + hi * 8);
        b.u[1] = ld16B(brow + k + 16 + hi * 8);
        acc = wmma_bf16(a, b, acc);
    }

#pragma unroll
    for (int r = 0; r < 8; ++r) {
        int row = mt * 16 + r + hi * 8;
        qout[(size_t)row * HDIM + ht * 16 + m16] = f2bf(acc[r]);
    }
}

// ---------------------------------------------------------------------------
// Phase 2: K = input @ Wk^T (stored [b,s,h]) and V = input @ Wv^T (stored
// transposed [b,h,s]).  One wave = one 16x16 (s,h) tile, both outputs,
// sharing the A-matrix fragments.  K = EMBED = 2048.
// ---------------------------------------------------------------------------
__global__ void __launch_bounds__(256)
kvproj_kernel(const unsigned short* __restrict__ inp,
              const unsigned short* __restrict__ wk,
              const unsigned short* __restrict__ wv,
              unsigned short* __restrict__ kout,
              unsigned short* __restrict__ vtout) {
    const int lane = threadIdx.x & 31;
    const int wave = threadIdx.x >> 5;
    const int tile = blockIdx.x * 8 + wave;         // 1024 tiles per batch
    const int b    = blockIdx.y;
    const int st = tile >> 3;                       // 0..127 (s tile)
    const int ht = tile & 7;                        // 0..7   (h tile)
    const int m16 = lane & 15;
    const int hi  = lane >> 4;

    v8f ak = {};
    v8f av = {};
    const unsigned short* arow  = inp + ((size_t)b * SEQ + st * 16 + m16) * EMBED;
    const unsigned short* bkrow = wk + (size_t)(ht * 16 + m16) * EMBED;
    const unsigned short* bvrow = wv + (size_t)(ht * 16 + m16) * EMBED;

    for (int k = 0; k < EMBED; k += 32) {
        FragBF a, bk, bv;
        a.u[0]  = ld16B(arow + k + hi * 8);
        a.u[1]  = ld16B(arow + k + 16 + hi * 8);
        bk.u[0] = ld16B(bkrow + k + hi * 8);
        bk.u[1] = ld16B(bkrow + k + 16 + hi * 8);
        bv.u[0] = ld16B(bvrow + k + hi * 8);
        bv.u[1] = ld16B(bvrow + k + 16 + hi * 8);
        ak = wmma_bf16(a, bk, ak);
        av = wmma_bf16(a, bv, av);
    }

#pragma unroll
    for (int r = 0; r < 8; ++r) {
        int s = st * 16 + r + hi * 8;
        int h = ht * 16 + m16;
        kout [((size_t)b * SEQ  + s) * HDIM + h] = f2bf(ak[r]);   // [b,s,h]
        vtout[((size_t)b * HDIM + h) * SEQ  + s] = f2bf(av[r]);   // [b,h,s]
    }
}

// ---------------------------------------------------------------------------
// Phase 3: causal flash attention.
// One block (8 waves) per (batch b, 16-query-row tile).  Waves split the
// causal S range in 32-column blocks, keep private fp32 accumulators +
// online-softmax stats, then merge via LDS log-sum-exp recombination.
// ---------------------------------------------------------------------------
__global__ void __launch_bounds__(256)
attn_kernel(const unsigned short* __restrict__ qbf,
            const unsigned short* __restrict__ kbf,
            const unsigned short* __restrict__ vtbf,
            float* __restrict__ out) {
    __shared__ __align__(16) unsigned short s_p[8][16 * 32]; // per-wave P tile
    __shared__ float s_m[8][16];
    __shared__ float s_l[8][16];
    __shared__ float s_acc[16 * HDIM];

    const int tid  = threadIdx.x;
    const int lane = tid & 31;
    const int wave = tid >> 5;
    const int m16  = lane & 15;
    const int hi   = lane >> 4;
    const int ntile = blockIdx.x;                 // 0..127
    const int b     = blockIdx.y;
    const int nbase = ntile * 16;

    for (int i = tid; i < 16 * HDIM; i += 256) s_acc[i] = 0.0f;

    // resident Q fragments (16 rows x 128 h), 4 K-chunks of 32
    FragBF aQ[4];
    const unsigned short* qrow = qbf + (size_t)(nbase + m16) * HDIM;
#pragma unroll
    for (int c = 0; c < 4; ++c) {
        aQ[c].u[0] = ld16B(qrow + c * 32 + hi * 8);
        aQ[c].u[1] = ld16B(qrow + c * 32 + 16 + hi * 8);
    }

    v8f O[8];
#pragma unroll
    for (int t = 0; t < 8; ++t) O[t] = (v8f){};
    float mrow[8], lrow[8];
#pragma unroll
    for (int r = 0; r < 8; ++r) { mrow[r] = -3.0e38f; lrow[r] = 0.0f; }

    const float kScale = 0.08838834764831845f * 1.44269504088896341f; // 1/sqrt(128)*log2(e)
    const int nb = (nbase + 15) / 32 + 1;         // causal 32-col blocks
    const unsigned short* kbase = kbf  + (size_t)b * SEQ  * HDIM;
    const unsigned short* vbase = vtbf + (size_t)b * HDIM * SEQ;

    for (int sb = wave; sb < nb; sb += 8) {
        const int s0 = sb * 32;
        v8f S0 = {};
        v8f S1 = {};
#pragma unroll
        for (int c = 0; c < 4; ++c) {
            FragBF bk;
            const unsigned short* kr0 =
                kbase + (size_t)(s0 + m16) * HDIM + c * 32 + hi * 8;
            bk.u[0] = ld16B(kr0);
            bk.u[1] = ld16B(kr0 + 16);
            S0 = wmma_bf16(aQ[c], bk, S0);
            const unsigned short* kr1 =
                kbase + (size_t)(s0 + 16 + m16) * HDIM + c * 32 + hi * 8;
            bk.u[0] = ld16B(kr1);
            bk.u[1] = ld16B(kr1 + 16);
            S1 = wmma_bf16(aQ[c], bk, S1);
        }

        // joint online softmax over the 32 columns
#pragma unroll
        for (int r = 0; r < 8; ++r) {
            const int qn = nbase + r + hi * 8;
            float x0 = S0[r] * kScale;
            if (s0 + m16 > qn) x0 = -1.0e30f;       // causal mask
            float x1 = S1[r] * kScale;
            if (s0 + 16 + m16 > qn) x1 = -1.0e30f;

            float red = fmaxf(x0, x1);              // row max across 16 lanes
            red = fmaxf(red, __shfl_xor(red, 1, 32));
            red = fmaxf(red, __shfl_xor(red, 2, 32));
            red = fmaxf(red, __shfl_xor(red, 4, 32));
            red = fmaxf(red, __shfl_xor(red, 8, 32));
            const float mnew  = fmaxf(mrow[r], red);
            const float alpha = __builtin_amdgcn_exp2f(mrow[r] - mnew);
            const float p0 = __builtin_amdgcn_exp2f(x0 - mnew);
            const float p1 = __builtin_amdgcn_exp2f(x1 - mnew);
            float ps = p0 + p1;                     // row sum across 16 lanes
            ps += __shfl_xor(ps, 1, 32);
            ps += __shfl_xor(ps, 2, 32);
            ps += __shfl_xor(ps, 4, 32);
            ps += __shfl_xor(ps, 8, 32);
            lrow[r] = lrow[r] * alpha + ps;
            mrow[r] = mnew;
#pragma unroll
            for (int t = 0; t < 8; ++t) O[t][r] *= alpha;

            const int prow = r + hi * 8;            // C-layout -> LDS (row-major)
            s_p[wave][prow * 32 + m16]      = f2bf(p0);
            s_p[wave][prow * 32 + 16 + m16] = f2bf(p1);
        }

        // order per-wave LDS stores before the transposed A-frag reads
        asm volatile("s_wait_dscnt 0" ::: "memory");

        FragBF aP;                                  // P as 16x32 bf16 A-frag
        {
            const unsigned short* pr = &s_p[wave][m16 * 32 + hi * 8];
            aP.u[0] = *reinterpret_cast<const v4u*>(pr);
            aP.u[1] = *reinterpret_cast<const v4u*>(pr + 16);
        }

#pragma unroll
        for (int t = 0; t < 8; ++t) {
            FragBF bv;
            const unsigned short* vr =
                vbase + (size_t)(t * 16 + m16) * SEQ + s0 + hi * 8;
            bv.u[0] = ld16B(vr);
            bv.u[1] = ld16B(vr + 16);
            O[t] = wmma_bf16(aP, bv, O[t]);
        }
    }

    // publish per-wave softmax stats
#pragma unroll
    for (int r = 0; r < 8; ++r) {
        const int row = r + hi * 8;
        s_m[wave][row] = mrow[r];
        s_l[wave][row] = lrow[r];
    }
    __syncthreads();

    // log-sum-exp recombination factors
    float fac[8];
#pragma unroll
    for (int r = 0; r < 8; ++r) {
        const int row = r + hi * 8;
        float M = -3.0e38f;
#pragma unroll
        for (int w = 0; w < 8; ++w) M = fmaxf(M, s_m[w][row]);
        float L = 0.0f;
#pragma unroll
        for (int w = 0; w < 8; ++w)
            L += __builtin_amdgcn_exp2f(s_m[w][row] - M) * s_l[w][row];
        fac[r] = __builtin_amdgcn_exp2f(mrow[r] - M) / L;
    }

    // accumulate partials (ds_add_f32)
#pragma unroll
    for (int t = 0; t < 8; ++t) {
#pragma unroll
        for (int r = 0; r < 8; ++r) {
            const int row = r + hi * 8;
            atomicAdd(&s_acc[row * HDIM + t * 16 + m16], O[t][r] * fac[r]);
        }
    }
    __syncthreads();

    float* orow = out + ((size_t)b * NLAT + nbase) * HDIM;
    for (int i = tid; i < 16 * HDIM; i += 256) orow[i] = s_acc[i];
}

// ---------------------------------------------------------------------------
// Host launcher
// ---------------------------------------------------------------------------
extern "C" void kernel_launch(void* const* d_in, const int* in_sizes, int n_in,
                              void* d_out, int out_size, void* d_ws, size_t ws_size,
                              hipStream_t stream) {
    (void)in_sizes; (void)n_in; (void)out_size; (void)ws_size;

    const float* input   = (const float*)d_in[0]; // [4,2048,2048]
    const float* latents = (const float*)d_in[1]; // [2048,512]
    const float* Wq      = (const float*)d_in[2]; // [128,512]
    const float* Wk      = (const float*)d_in[3]; // [128,2048]
    const float* Wv      = (const float*)d_in[4]; // [128,2048]
    float* out           = (float*)d_out;         // [4,2048,128]

    char* ws = (char*)d_ws;
    // byte offsets (each region size is a multiple of 256)
    const size_t OFF_IN  = 0;                         // 4*2048*2048*2 = 32 MB
    const size_t OFF_LAT = OFF_IN  + (size_t)NBATCH * SEQ * EMBED * 2;
    const size_t OFF_WQ  = OFF_LAT + (size_t)NLAT * LDIM * 2;
    const size_t OFF_WK  = OFF_WQ  + (size_t)HDIM * LDIM * 2;
    const size_t OFF_WV  = OFF_WK  + (size_t)HDIM * EMBED * 2;
    const size_t OFF_Q   = OFF_WV  + (size_t)HDIM * EMBED * 2;
    const size_t OFF_K   = OFF_Q   + (size_t)NLAT * HDIM * 2;
    const size_t OFF_VT  = OFF_K   + (size_t)NBATCH * SEQ * HDIM * 2;

    unsigned short* in_bf  = (unsigned short*)(ws + OFF_IN);
    unsigned short* lat_bf = (unsigned short*)(ws + OFF_LAT);
    unsigned short* wq_bf  = (unsigned short*)(ws + OFF_WQ);
    unsigned short* wk_bf  = (unsigned short*)(ws + OFF_WK);
    unsigned short* wv_bf  = (unsigned short*)(ws + OFF_WV);
    unsigned short* q_bf   = (unsigned short*)(ws + OFF_Q);
    unsigned short* k_bf   = (unsigned short*)(ws + OFF_K);
    unsigned short* vt_bf  = (unsigned short*)(ws + OFF_VT);

    // Phase 0: convert all fp32 inputs to bf16
    {
        int np;
        np = NBATCH * SEQ * EMBED / 2;
        cvt_bf16_kernel<<<(np + 255) / 256, 256, 0, stream>>>(input,   (unsigned int*)in_bf,  np);
        np = NLAT * LDIM / 2;
        cvt_bf16_kernel<<<(np + 255) / 256, 256, 0, stream>>>(latents, (unsigned int*)lat_bf, np);
        np = HDIM * LDIM / 2;
        cvt_bf16_kernel<<<(np + 255) / 256, 256, 0, stream>>>(Wq,      (unsigned int*)wq_bf,  np);
        np = HDIM * EMBED / 2;
        cvt_bf16_kernel<<<(np + 255) / 256, 256, 0, stream>>>(Wk,      (unsigned int*)wk_bf,  np);
        cvt_bf16_kernel<<<(np + 255) / 256, 256, 0, stream>>>(Wv,      (unsigned int*)wv_bf,  np);
    }

    // Phase 1: Q projection  (128 blocks x 8 waves = 1024 16x16 tiles)
    qproj_kernel<<<dim3(128), 256, 0, stream>>>(lat_bf, wq_bf, q_bf);

    // Phase 2: K/V projections (per batch: 1024 tiles)
    kvproj_kernel<<<dim3(128, NBATCH), 256, 0, stream>>>(in_bf, wk_bf, wv_bf, k_bf, vt_bf);

    // Phase 3: causal flash attention (one block per 16 query rows per batch)
    attn_kernel<<<dim3(NLAT / 16, NBATCH), 256, 0, stream>>>(q_bf, k_bf, vt_bf, out);
}
